// SSE_50491635531784
// MI455X (gfx1250) — compile-verified, matrix operand order checked by reading, not run on hardware
//
#include <hip/hip_runtime.h>

#define N_UTT 2048
#define DIM   768
#define SCALE 0.03608439182435161f   // 1/sqrt(768)

typedef __attribute__((ext_vector_type(16))) __bf16 v16bf;
typedef __attribute__((ext_vector_type(8)))  float  v8f;

__device__ __forceinline__ unsigned short f2bf(float f) {
  unsigned int u = __float_as_uint(f);
  u += 0x7FFFu + ((u >> 16) & 1u);            // round-to-nearest-even
  return (unsigned short)(u >> 16);
}
__device__ __forceinline__ float bf2f(unsigned short u) {
  return __uint_as_float(((unsigned int)u) << 16);
}

// ---------------- conversion kernels ----------------
__global__ void k_f32_to_bf16(const float* __restrict__ src,
                              unsigned short* __restrict__ dst, int n) {
  int i = blockIdx.x * 256 + threadIdx.x;
  if (i < n) dst[i] = f2bf(src[i]);
}
// dst[r*cols+c] = src[r*ld + coff + c]   (extract Wb from W_intra)
__global__ void k_f32_to_bf16_slice(const float* __restrict__ src,
                                    unsigned short* __restrict__ dst,
                                    int rows, int cols, int ld, int coff) {
  int i = blockIdx.x * 256 + threadIdx.x;
  if (i < rows * cols) {
    int r = i / cols, c = i % cols;
    dst[i] = f2bf(src[(size_t)r * ld + coff + c]);
  }
}
// dst[c*rows + r] = src[r*ld + coff + c]  (transposed extract: WaT[n][k] = Wa[k][n])
__global__ void k_f32_to_bf16_transpose(const float* __restrict__ src,
                                        unsigned short* __restrict__ dst,
                                        int rows, int cols, int ld, int coff) {
  int i = blockIdx.x * 256 + threadIdx.x;
  if (i < rows * cols) {
    int c = i / rows, r = i % rows;        // dst row-major [cols][rows]
    dst[i] = f2bf(src[(size_t)r * ld + coff + c]);
  }
}

// ---------------- WMMA bf16 GEMM:  out[m][n] = sum_k A[m][k]*W[n][k] (+bias[n]) ----
// One wave per 16x32 output tile (A fragment reused across two B fragments).
// K multiple of 32, M multiple of 16, Ncols multiple of 32.
__global__ __launch_bounds__(128)
void k_gemm_wmma_bf16(const unsigned short* __restrict__ A,
                      const unsigned short* __restrict__ W,
                      const float* __restrict__ bias,
                      float* __restrict__ outF,
                      unsigned short* __restrict__ outB,
                      int M, int Ncols, int K) {
  const int tilesN = Ncols >> 5;
  const int tile = blockIdx.x * 4 + (threadIdx.x >> 5);
  if (tile >= (M >> 4) * tilesN) return;      // whole-wave exit: EXEC stays all-1 for WMMA
  const int m0 = (tile / tilesN) << 4;
  const int n0 = (tile % tilesN) << 5;
  const int lane = threadIdx.x & 31;
  const int lo16 = lane & 15;
  const int kb = (lane < 16) ? 0 : 8;         // ISA 16-bit A/B fragment K base per lane half

  union V16 { uint4 q[2]; v16bf v; };

  v8f acc0 = {}, acc1 = {};
  const unsigned short* arow  = A + (size_t)(m0 + lo16) * K + kb;
  const unsigned short* brow0 = W + (size_t)(n0 + lo16) * K + kb;
  const unsigned short* brow1 = W + (size_t)(n0 + 16 + lo16) * K + kb;
#pragma unroll 2
  for (int k0 = 0; k0 < K; k0 += 32) {
    V16 a, b0, b1;
    a.q[0]  = *(const uint4*)(arow  + k0);        // elems 0..7  : K = kb+0..7
    a.q[1]  = *(const uint4*)(arow  + k0 + 16);   // elems 8..15 : K = kb+16..23
    b0.q[0] = *(const uint4*)(brow0 + k0);
    b0.q[1] = *(const uint4*)(brow0 + k0 + 16);
    b1.q[0] = *(const uint4*)(brow1 + k0);
    b1.q[1] = *(const uint4*)(brow1 + k0 + 16);
    acc0 = __builtin_amdgcn_wmma_f32_16x16x32_bf16(false, a.v, false, b0.v,
                                                   (short)0, acc0, false, false);
    acc1 = __builtin_amdgcn_wmma_f32_16x16x32_bf16(false, a.v, false, b1.v,
                                                   (short)0, acc1, false, false);
  }
#pragma unroll
  for (int j = 0; j < 8; ++j) {
    int m = m0 + j + ((lane >> 4) << 3);      // C/D layout: VGPR j -> row, lane%16 -> col
    int n = n0 + lo16;
    float v0 = acc0[j] + (bias ? bias[n] : 0.0f);
    float v1 = acc1[j] + (bias ? bias[n + 16] : 0.0f);
    if (outF) {
      outF[(size_t)m * Ncols + n]      = v0;
      outF[(size_t)m * Ncols + n + 16] = v1;
    } else {
      outB[(size_t)m * Ncols + n]      = f2bf(v0);
      outB[(size_t)m * Ncols + n + 16] = f2bf(v1);
    }
  }
}

// ---------------- sequential scan (1 block, 32 waves) ----------------
// matvec: out[r] = sum_k bf16(W[r][k]) * x[k] + bias[r], r in [0,DIM)
__device__ __forceinline__ void matvec768(const unsigned short* __restrict__ W,
                                          const float* __restrict__ x,
                                          const float* __restrict__ bias,
                                          float* __restrict__ out,
                                          int wid, int lane) {
  for (int r = wid; r < DIM; r += 32) {
    const unsigned short* wr = W + (size_t)r * DIM;
    float s = 0.f;
    for (int k = 2 * lane; k < DIM; k += 64) {        // packed pair of bf16 per b32 load
      unsigned int w = *(const unsigned int*)(wr + k);
      s = fmaf(bf2f((unsigned short)w), x[k], s);
      s = fmaf(bf2f((unsigned short)(w >> 16)), x[k + 1], s);
    }
#pragma unroll
    for (int m = 16; m; m >>= 1) s += __shfl_xor(s, m, 32);
    if (lane == 0) out[r] = s + bias[r];
  }
}

// masked softmax-attention over rows [lo,hi] of Kmat/Vmat with query q;
// accumulates result into vacc[0..DIM).
__device__ __forceinline__ void attend(const float* __restrict__ Kmat,
                                       const float* __restrict__ Vmat,
                                       const float* __restrict__ q,
                                       int lo, int hi,
                                       float* sc, float* colred, float* red, float* bcast,
                                       float* vacc, int t, int wid, int lane) {
  __syncthreads();                              // q / previous phase visible
  const float4* q4 = (const float4*)q;
  // scores: one wave per row, float4 loads (6 iters of b128)
  for (int r = lo + wid; r <= hi; r += 32) {
    const float4* kr4 = (const float4*)(Kmat + (size_t)r * DIM);
    if (r + 32 <= hi) __builtin_prefetch(Kmat + (size_t)(r + 32) * DIM, 0, 1);
    float s = 0.f;
    for (int k = lane; k < DIM / 4; k += 32) {
      float4 a = kr4[k], b = q4[k];
      s = fmaf(a.x, b.x, s); s = fmaf(a.y, b.y, s);
      s = fmaf(a.z, b.z, s); s = fmaf(a.w, b.w, s);
    }
#pragma unroll
    for (int m = 16; m; m >>= 1) s += __shfl_xor(s, m, 32);
    if (lane == 0) sc[r] = s * SCALE;
  }
  __syncthreads();
  // block max
  float mx = -3.4e38f;
  for (int r = lo + t; r <= hi; r += 1024) mx = fmaxf(mx, sc[r]);
#pragma unroll
  for (int m = 16; m; m >>= 1) mx = fmaxf(mx, __shfl_xor(mx, m, 32));
  if (lane == 0) red[wid] = mx;
  __syncthreads();
  if (wid == 0) {
    float v = red[lane];
#pragma unroll
    for (int m = 16; m; m >>= 1) v = fmaxf(v, __shfl_xor(v, m, 32));
    if (lane == 0) *bcast = v;
  }
  __syncthreads();
  mx = *bcast;
  // exp + block sum
  float sum = 0.f;
  for (int r = lo + t; r <= hi; r += 1024) { float e = __expf(sc[r] - mx); sc[r] = e; sum += e; }
#pragma unroll
  for (int m = 16; m; m >>= 1) sum += __shfl_xor(sum, m, 32);
  if (lane == 0) red[wid] = sum;
  __syncthreads();
  if (wid == 0) {
    float v = red[lane];
#pragma unroll
    for (int m = 16; m; m >>= 1) v += __shfl_xor(v, m, 32);
    if (lane == 0) *bcast = v;
  }
  __syncthreads();
  const float inv = 1.f / *bcast;
  // vacc[c] += inv * sum_r sc[r] * Vmat[r][c]   (4 row-groups x 256 columns)
  const int g = t >> 8, cl = t & 255;
  for (int cb = 0; cb < DIM; cb += 256) {
    int c = cb + cl;
    float p = 0.f;
    for (int r = lo + g; r <= hi; r += 4) p = fmaf(sc[r], Vmat[(size_t)r * DIM + c], p);
    colred[t] = p;
    __syncthreads();
    if (g == 0)
      vacc[c] += (colred[cl] + colred[256 + cl] + colred[512 + cl] + colred[768 + cl]) * inv;
    __syncthreads();
  }
}

__global__ __launch_bounds__(1024)
void k_scan(const float* __restrict__ U, const int* __restrict__ speaker,
            const float* __restrict__ K1, const float* __restrict__ V1,
            const float* __restrict__ Q2, const float* __restrict__ Cpart,
            const unsigned short* __restrict__ Wcb,
            const unsigned short* __restrict__ Wk2b,
            const unsigned short* __restrict__ Wv2b,
            const float* __restrict__ bk2, const float* __restrict__ bv2,
            float* __restrict__ V, float* __restrict__ K2, float* __restrict__ V2) {
  __shared__ __align__(16) float q1s[DIM];
  __shared__ __align__(16) float q2s[DIM];
  __shared__ __align__(16) float xv[DIM];
  __shared__ __align__(16) float vacc[DIM];
  __shared__ __align__(16) float sc[N_UTT];
  __shared__ __align__(16) float colred[1024];
  __shared__ float red[32];
  __shared__ float bcast;
  __shared__ int   lastIdx[16];
  __shared__ int   s_tmp;

  const int t = threadIdx.x;
  const int wid = t >> 5, lane = t & 31;
  if (t < 16) lastIdx[t] = -1;
  __syncthreads();

  for (int i = 0; i < N_UTT; ++i) {
    if (t == 0) s_tmp = lastIdx[speaker[i]];
    __syncthreads();
    const int tmp = s_tmp;

    if (tmp >= 0) {
      for (int c = t; c < DIM; c += 1024) {
        xv[c] = V[(size_t)tmp * DIM + c];      // v_prev
        vacc[c] = 0.f;
        q2s[c] = Q2[(size_t)i * DIM + c];      // precomputed inter query
      }
      __syncthreads();
      matvec768(Wcb, xv, Cpart + (size_t)i * DIM, q1s, wid, lane);  // q1 = Wc*v_prev + Cpart[i]
      attend(K1, V1, q1s, 0, i, sc, colred, red, &bcast, vacc, t, wid, lane);       // intra
      attend(K2, V2, q2s, tmp, i - 1, sc, colred, red, &bcast, vacc, t, wid, lane); // inter
      __syncthreads();
      for (int c = t; c < DIM; c += 1024) {
        float v = vacc[c];
        V[(size_t)i * DIM + c] = v;
        xv[c] = v;
      }
    } else {
      for (int c = t; c < DIM; c += 1024) {
        float v = U[(size_t)i * DIM + c];
        V[(size_t)i * DIM + c] = v;
        xv[c] = v;
      }
    }
    __syncthreads();
    matvec768(Wk2b, xv, bk2, K2 + (size_t)i * DIM, wid, lane);   // extend inter keys
    matvec768(Wv2b, xv, bv2, V2 + (size_t)i * DIM, wid, lane);   // extend inter values
    if (t == 0) lastIdx[speaker[i]] = i;
    __threadfence_block();
    __syncthreads();
  }
}

// ---------------- host launch ----------------
extern "C" void kernel_launch(void* const* d_in, const int* in_sizes, int n_in,
                              void* d_out, int out_size, void* d_ws, size_t ws_size,
                              hipStream_t stream) {
  (void)in_sizes; (void)n_in; (void)out_size; (void)ws_size;
  const float* U       = (const float*)d_in[0];
  const int*   spk     = (const int*)d_in[1];
  const float* W_intra = (const float*)d_in[2];
  const float* b_intra = (const float*)d_in[3];
  const float* W_inter = (const float*)d_in[4];
  const float* b_inter = (const float*)d_in[5];
  const float* Wq1 = (const float*)d_in[6];  const float* bq1 = (const float*)d_in[7];
  const float* Wk1 = (const float*)d_in[8];  const float* bk1 = (const float*)d_in[9];
  const float* Wv1 = (const float*)d_in[10]; const float* bv1 = (const float*)d_in[11];
  const float* Wq2 = (const float*)d_in[12]; const float* bq2 = (const float*)d_in[13];
  const float* Wk2 = (const float*)d_in[14]; const float* bk2 = (const float*)d_in[15];
  const float* Wv2 = (const float*)d_in[16]; const float* bv2 = (const float*)d_in[17];

  char* base = (char*)d_ws;
  size_t off = 0;
  auto alloc = [&](size_t bytes) -> void* {
    void* p = base + off;
    off = (off + bytes + 255) & ~(size_t)255;
    return p;
  };
  const size_t ND = (size_t)N_UTT * DIM, DD = (size_t)DIM * DIM;
  unsigned short* Ub      = (unsigned short*)alloc(ND * 2);
  unsigned short* Wk1b    = (unsigned short*)alloc(DD * 2);
  unsigned short* Wv1b    = (unsigned short*)alloc(DD * 2);
  unsigned short* Winterb = (unsigned short*)alloc(DD * 2);
  unsigned short* Wq2b    = (unsigned short*)alloc(DD * 2);
  unsigned short* Wq1b    = (unsigned short*)alloc(DD * 2);
  unsigned short* WaTb    = (unsigned short*)alloc(DD * 2);   // WaT[n][k] = Wa[k][n]
  unsigned short* Wbb     = (unsigned short*)alloc(DD * 2);
  unsigned short* Wk2b    = (unsigned short*)alloc(DD * 2);
  unsigned short* Wv2b    = (unsigned short*)alloc(DD * 2);
  unsigned short* Tb      = (unsigned short*)alloc(ND * 2);
  unsigned short* T2b     = (unsigned short*)alloc(ND * 2);
  unsigned short* Wcb     = (unsigned short*)alloc(DD * 2);
  float* K1f = (float*)alloc(ND * 4);
  float* V1f = (float*)alloc(ND * 4);
  float* Q2f = (float*)alloc(ND * 4);
  float* Cpf = (float*)alloc(ND * 4);
  float* K2f = (float*)alloc(ND * 4);
  float* V2f = (float*)alloc(ND * 4);

  auto conv = [&](const float* s, unsigned short* d, int n) {
    k_f32_to_bf16<<<(n + 255) / 256, 256, 0, stream>>>(s, d, n);
  };
  // bf16 copies of everything the GEMMs touch
  conv(U, Ub, (int)ND);
  conv(Wk1, Wk1b, (int)DD);  conv(Wv1, Wv1b, (int)DD);
  conv(W_inter, Winterb, (int)DD); conv(Wq2, Wq2b, (int)DD);
  conv(Wq1, Wq1b, (int)DD);  conv(Wk2, Wk2b, (int)DD);  conv(Wv2, Wv2b, (int)DD);
  k_f32_to_bf16_transpose<<<((int)DD + 255) / 256, 256, 0, stream>>>(W_intra, WaTb, DIM, DIM, 2 * DIM, 0);
  k_f32_to_bf16_slice<<<((int)DD + 255) / 256, 256, 0, stream>>>(W_intra, Wbb, DIM, DIM, 2 * DIM, DIM);

  auto gemm = [&](const unsigned short* A, const unsigned short* W, const float* bias,
                  float* oF, unsigned short* oB, int M) {
    int tiles = (M / 16) * (DIM / 32);
    k_gemm_wmma_bf16<<<(tiles + 3) / 4, 128, 0, stream>>>(A, W, bias, oF, oB, M, DIM, DIM);
  };
  gemm(Ub, Wk1b, bk1, K1f, nullptr, N_UTT);               // K1 = U Wk1^T + bk1
  gemm(Ub, Wv1b, bv1, V1f, nullptr, N_UTT);               // V1 = U Wv1^T + bv1
  gemm(Ub, Winterb, b_inter, nullptr, Tb, N_UTT);         // T  = U W_inter^T + b_inter
  gemm(Ub, Wbb, b_intra, nullptr, T2b, N_UTT);            // T2 = U Wb^T + b_intra
  gemm(Tb, Wq2b, bq2, Q2f, nullptr, N_UTT);               // Q2 = T Wq2^T + bq2
  gemm(T2b, Wq1b, bq1, Cpf, nullptr, N_UTT);              // Cpart = T2 Wq1^T + bq1
  gemm(Wq1b, WaTb, nullptr, nullptr, Wcb, DIM);           // Wc = Wq1 @ Wa  (via WaT)

  k_scan<<<1, 1024, 0, stream>>>(U, spk, K1f, V1f, Q2f, Cpf,
                                 Wcb, Wk2b, Wv2b, bk2, bv2,
                                 (float*)d_out, K2f, V2f);
}
